// Takahashi_40716289966842
// MI455X (gfx1250) — matrix operand order repaired and checked by reading, not compiled, not run on hardware
//
#include <hip/hip_runtime.h>
#include <hip/hip_bf16.h>
#include <math.h>

typedef __attribute__((ext_vector_type(16))) _Float16 v16h;
typedef __attribute__((ext_vector_type(8)))  float    v8f;

#define NS_   4
#define NMAX_ 8
#define LMAX_ 4
#define NO_   20
#define HID_  128
#define D_    32      // NMAX_*LMAX_
#define RCUT_ 6.0f

// -------------------------------------------------------------------------
// WMMA fragment layouts used here (wave32; A/B K order is contiguous, not the ISA table order):
// A 16x32 f16: grp=lane>>4, m=lane&15; elem i -> K = 16*grp + i
// B 32x16 f16: n = lane&15; kb = (lane>>4)*16; elem i -> K = kb+i
// C/D 16x16 f32: elem r -> row r + 8*grp, col = lane&15
// -------------------------------------------------------------------------
__device__ __forceinline__ v16h load_a_frag(const _Float16* src, int stride, int lane) {
    int grp = lane >> 4, m = lane & 15;
    v16h A;
#pragma unroll
    for (int i = 0; i < 16; ++i) {
        int k = i + 16 * grp;
        A[i] = src[m * stride + k];
    }
    return A;
}

// -------------------------------------------------------------------------
// kernel 0: zero workspace (float4)
// -------------------------------------------------------------------------
__global__ void zero_f4(float4* __restrict__ p, int n4) {
    int i = blockIdx.x * blockDim.x + threadIdx.x;
    if (i < n4) p[i] = make_float4(0.f, 0.f, 0.f, 0.f);
}

// -------------------------------------------------------------------------
// kernel 1: cn0[a, f] = alpha[symbols[a], f]
// -------------------------------------------------------------------------
__global__ void gather_alpha(const int* __restrict__ sym, const float* __restrict__ alpha,
                             float* __restrict__ cn0, int nta) {
    int i = blockIdx.x * blockDim.x + threadIdx.x;
    if (i < nta * D_) {
        int a = i / D_, f = i % D_;
        cn0[i] = alpha[sym[a] * D_ + f];
    }
}

// -------------------------------------------------------------------------
// kernel 1b: convert W1/W2 to f16 pre-swizzled into B-fragment lane layout.
// pW1 fragment id f = s*8+nt      (cols nt*16..nt*16+15, K = 0..31)
// pW2 fragment id f = s*8+kk*2+nt (cols nt*16..+15,      K = kk*32..+31)
// Each fragment: 32 lanes x 16 halves contiguous -> one 32B vector load/lane.
// -------------------------------------------------------------------------
__global__ void prep_weights(const float* __restrict__ W1, const float* __restrict__ W2,
                             _Float16* __restrict__ pW1, _Float16* __restrict__ pW2) {
    int t = blockIdx.x * blockDim.x + threadIdx.x;
    if (t >= NS_ * 8 * 32) return;
    int lane = t & 31;
    int frag = t >> 5;            // 0..31
    int s = frag >> 3, nt = frag & 7;
    int n = lane & 15, kb = (lane >> 4) * 16;
    int kk = nt >> 1, n2 = nt & 1;   // W2 decomposition of the same frag id
    v16h a, b;
#pragma unroll
    for (int i = 0; i < 16; ++i) {
        a[i] = (_Float16)W1[(size_t)s * D_ * HID_ + (kb + i) * HID_ + nt * 16 + n];
        b[i] = (_Float16)W2[(size_t)s * HID_ * D_ + (kk * 32 + kb + i) * D_ + n2 * 16 + n];
    }
    ((v16h*)pW1)[frag * 32 + lane] = a;
    ((v16h*)pW2)[frag * 32 + lane] = b;
}

// -------------------------------------------------------------------------
// kernel 2: edge scatter.
// dens[i, n, o] += fcut(d) * ux^p uy^q uz^r * cn[j, n, L(o)]
// -------------------------------------------------------------------------
__global__ __launch_bounds__(256) void edge_scatter(
    const int* __restrict__ iidx, const int* __restrict__ jidx,
    const float* __restrict__ disp, const float* __restrict__ dist,
    const float* __restrict__ cn, float* __restrict__ dens, int nn) {
    int e = blockIdx.x * blockDim.x + threadIdx.x;
    if (e >= nn) return;
    int i = iidx[e], j = jidx[e];
    float d  = dist[e];
    float inv = 1.0f / d;
    float ux = disp[e * 3 + 0] * inv;
    float uy = disp[e * 3 + 1] * inv;
    float uz = disp[e * 3 + 2] * inv;
    float c  = __cosf(d * (3.14159265358979f / RCUT_));
    float fc = 0.25f * (c + 1.0f) * (c + 1.0f);

    float xp[4] = {1.f, ux, ux * ux, ux * ux * ux};
    float yq[4] = {1.f, uy, uy * uy, uy * uy * uy};
    float zr[4] = {1.f, uz, uz * uz, uz * uz * uz};

    const int P[NO_]  = {0, 0,0,1, 0,0,0,1,1,2, 0,0,0,0,1,1,1,2,2,3};
    const int Q[NO_]  = {0, 0,1,0, 0,1,2,0,1,0, 0,1,2,3,0,1,2,0,1,0};
    const int R[NO_]  = {0, 1,0,0, 2,1,0,1,0,0, 3,2,1,0,2,1,0,1,0,0};
    const int LO[NO_] = {0, 1,1,1, 2,2,2,2,2,2, 3,3,3,3,3,3,3,3,3,3};

    float cnj[D_];
#pragma unroll
    for (int f = 0; f < D_; ++f) cnj[f] = cn[(size_t)j * D_ + f];

    float* db = dens + (size_t)i * (NMAX_ * NO_);
#pragma unroll
    for (int o = 0; o < NO_; ++o) {
        float a = fc * xp[P[o]] * yq[Q[o]] * zr[R[o]];
        int l = LO[o];
#pragma unroll
        for (int n = 0; n < NMAX_; ++n) {
            unsafeAtomicAdd(&db[n * NO_ + o], a * cnj[n * LMAX_ + l]);
        }
    }
}

// -------------------------------------------------------------------------
// kernel 3: bnl[a, n, l] = hop_rowsum[n,l] * sum_{o: L(o)=l} dens[a,n,o]^2
// (TLM_ROWSUM == 1: Chebyshev row sums = T_l(1) = 1)
// -------------------------------------------------------------------------
__global__ void bnl_reduce(const float* __restrict__ dens,
                           const float* __restrict__ hop,  // pass base: [NMAX][LMAX][LMAX]
                           float* __restrict__ out, int nta) {
    int t = blockIdx.x * blockDim.x + threadIdx.x;
    if (t >= nta * NMAX_) return;
    int a = t / NMAX_, n = t % NMAX_;
    const int LO[NO_] = {0, 1,1,1, 2,2,2,2,2,2, 3,3,3,3,3,3,3,3,3,3};
    const float* dp = dens + (size_t)t * NO_;
    float s[LMAX_] = {0.f, 0.f, 0.f, 0.f};
#pragma unroll
    for (int o = 0; o < NO_; ++o) {
        float v = dp[o];
        s[LO[o]] += v * v;
    }
#pragma unroll
    for (int l = 0; l < LMAX_; ++l) {
        const float* hr = hop + (n * LMAX_ + l) * LMAX_;
        float hrs = hr[0] + hr[1] + hr[2] + hr[3];
        out[(size_t)a * D_ + n * LMAX_ + l] = s[l] * hrs;
    }
}

// -------------------------------------------------------------------------
// kernel 4: species-routed MLP via v_wmma_f32_16x16x32_f16.
// One wave per 16-atom tile; B operands come pre-swizzled (one 32B load/lane).
// -------------------------------------------------------------------------
__global__ __launch_bounds__(32) void mlp_wmma(
    const float* __restrict__ xfeat,
    const v16h* __restrict__ pW1, const v16h* __restrict__ pW2,
    const float* __restrict__ b1, const float* __restrict__ b2,
    const int* __restrict__ sym,
    const float* __restrict__ cn_in, float* __restrict__ cn_out, int nta) {
    __shared__ _Float16 Xs[16][D_];
    __shared__ _Float16 Hs[16][HID_];
    int lane = threadIdx.x;
    int a0 = blockIdx.x * 16;

    for (int t = lane; t < 16 * D_; t += 32) {
        int m = t >> 5, f = t & 31;
        int a = a0 + m;
        Xs[m][f] = (_Float16)(a < nta ? xfeat[(size_t)a * D_ + f] : 0.0f);
    }
    __syncthreads();

    v16h Af = load_a_frag(&Xs[0][0], D_, lane);
    int grp = lane >> 4, nc = lane & 15;
    const v8f vz = {0.f, 0.f, 0.f, 0.f, 0.f, 0.f, 0.f, 0.f};

    for (int s = 0; s < NS_; ++s) {
        // ---- GEMM 1: h = x @ W1[s] ----
        v8f h[8];
#pragma unroll
        for (int nt = 0; nt < 8; ++nt) {
            v16h Bf = pW1[(s * 8 + nt) * 32 + lane];
            h[nt] = __builtin_amdgcn_wmma_f32_16x16x32_f16(
                false, Af, false, Bf, (short)0, vz, false, false);
        }
        // bias + leaky_relu, store h to LDS in D-fragment layout
#pragma unroll
        for (int nt = 0; nt < 8; ++nt) {
#pragma unroll
            for (int r = 0; r < 8; ++r) {
                int col = nt * 16 + nc, row = r + 8 * grp;
                float v = h[nt][r] + b1[s * HID_ + col];
                v = v > 0.f ? v : 0.01f * v;
                Hs[row][col] = (_Float16)v;
            }
        }
        __syncthreads();

        // ---- GEMM 2: y = h @ W2[s] ----
        v8f y0 = vz, y1 = vz;
#pragma unroll
        for (int kk = 0; kk < 4; ++kk) {
            v16h A2  = load_a_frag(&Hs[0][kk * 32], HID_, lane);
            v16h B20 = pW2[(s * 8 + kk * 2 + 0) * 32 + lane];
            v16h B21 = pW2[(s * 8 + kk * 2 + 1) * 32 + lane];
            y0 = __builtin_amdgcn_wmma_f32_16x16x32_f16(
                false, A2, false, B20, (short)0, y0, false, false);
            y1 = __builtin_amdgcn_wmma_f32_16x16x32_f16(
                false, A2, false, B21, (short)0, y1, false, false);
        }
        // ---- routed update: cn_out = cn_in + y for atoms of species s ----
#pragma unroll
        for (int r = 0; r < 8; ++r) {
            int row = r + 8 * grp;
            int a = a0 + row;
            if (a < nta && sym[a] == s) {
                int c0 = nc, c1 = 16 + nc;
                cn_out[(size_t)a * D_ + c0] = cn_in[(size_t)a * D_ + c0] + (y0[r] + b2[s * D_ + c0]);
                cn_out[(size_t)a * D_ + c1] = cn_in[(size_t)a * D_ + c1] + (y1[r] + b2[s * D_ + c1]);
            }
        }
        __syncthreads();  // Hs reused by next species
    }
}

// -------------------------------------------------------------------------
// host
// -------------------------------------------------------------------------
extern "C" void kernel_launch(void* const* d_in, const int* in_sizes, int n_in,
                              void* d_out, int out_size, void* d_ws, size_t ws_size,
                              hipStream_t stream) {
    const int*   symbols = (const int*)d_in[0];
    const int*   iidx    = (const int*)d_in[1];
    const int*   jidx    = (const int*)d_in[2];
    const float* disp    = (const float*)d_in[3];
    const float* dist    = (const float*)d_in[4];
    const float* alpha   = (const float*)d_in[5];
    const float* hop     = (const float*)d_in[6];   // (2, 8, 4, 4)
    const float* W1      = (const float*)d_in[7];   // (1, 4, 32, 128)
    const float* b1      = (const float*)d_in[8];   // (1, 4, 128)
    const float* W2      = (const float*)d_in[9];   // (1, 4, 128, 32)
    const float* b2      = (const float*)d_in[10];  // (1, 4, 32)
    float* out = (float*)d_out;

    int nta = in_sizes[0];
    int nn  = in_sizes[1];

    char*  ws    = (char*)d_ws;
    size_t densB = (size_t)nta * NMAX_ * NO_ * sizeof(float);   // 16 MB, 32B-aligned
    float* dens  = (float*)ws;
    float* cn0   = (float*)(ws + densB);
    float* cn1   = cn0 + (size_t)nta * D_;
    float* xfeat = cn1 + (size_t)nta * D_;
    _Float16* pW1 = (_Float16*)(xfeat + (size_t)nta * D_);      // 32 KB
    _Float16* pW2 = pW1 + NS_ * 8 * 32 * 16;                    // 32 KB

    int densN  = nta * NMAX_ * NO_;
    int densN4 = densN / 4;
    int edgeBlocks = (nn + 255) / 256;
    int tileBlocks = (nta + 15) / 16;

    // one-time per launch: cn0 gather + weight swizzle (both tiny, L2-resident)
    gather_alpha<<<(nta * D_ + 255) / 256, 256, 0, stream>>>(symbols, alpha, cn0, nta);
    prep_weights<<<4, 256, 0, stream>>>(W1, W2, pW1, pW2);

    // pass 0
    zero_f4<<<(densN4 + 255) / 256, 256, 0, stream>>>((float4*)dens, densN4);
    edge_scatter<<<edgeBlocks, 256, 0, stream>>>(iidx, jidx, disp, dist, cn0, dens, nn);
    bnl_reduce<<<(nta * NMAX_ + 255) / 256, 256, 0, stream>>>(dens, hop, xfeat, nta);

    // MLP: cn1 = cn0 + G(xfeat)
    mlp_wmma<<<tileBlocks, 32, 0, stream>>>(xfeat, (const v16h*)pW1, (const v16h*)pW2,
                                            b1, b2, symbols, cn0, cn1, nta);

    // pass 1 -> output
    zero_f4<<<(densN4 + 255) / 256, 256, 0, stream>>>((float4*)dens, densN4);
    edge_scatter<<<edgeBlocks, 256, 0, stream>>>(iidx, jidx, disp, dist, cn1, dens, nn);
    bnl_reduce<<<(nta * NMAX_ + 255) / 256, 256, 0, stream>>>(
        dens, hop + NMAX_ * LMAX_ * LMAX_, out, nta);
}
